// MonteCarloBlock_56444460204082
// MI455X (gfx1250) — matrix operand order repaired and checked by reading, not compiled, not run on hardware
//
#include <hip/hip_runtime.h>
#include <hip/hip_bf16.h>

// ---------------------------------------------------------------------------
// Problem constants (from reference): B=512 graphs, N=64 nodes, Fin=128,
// H=256 hidden, K=32 clusters, S_MC=4, TEMP=0.1
// ---------------------------------------------------------------------------
#define NB   512
#define NN   64
#define FIN  128
#define HH   256
#define KC   32
#define SMC  4

typedef __attribute__((ext_vector_type(8)))  _Float16 v8h;
typedef __attribute__((ext_vector_type(16))) _Float16 v16h;
typedef __attribute__((ext_vector_type(8)))  float    v8f;

// A-fragment: 16x32 f16 tile, row-major [M][K]. Lane l (lr=l&15, lh=l>>4):
// row = tile_m*16+lr, 16 contiguous halves starting at k32+lh*16.
static __device__ inline v16h afrag(const _Float16* base, int row, int rowlen,
                                    int k32, int lh) {
  const _Float16* p = base + (size_t)row * rowlen + k32 + lh * 16;
  v8h lo = *(const v8h*)p;
  v8h hi = *(const v8h*)(p + 8);
  return __builtin_shufflevector(lo, hi, 0,1,2,3,4,5,6,7,8,9,10,11,12,13,14,15);
}

// B-fragment from B^T storage ([N][K] row-major): lane l holds col n = tile_n*16
// + (l&15), 16 contiguous halves starting at k32 + (l>>4)*16.  32B aligned.
static __device__ inline v16h bfrag(const _Float16* base, int n, int rowlen,
                                    int k32, int lh) {
  const _Float16* p = base + (size_t)n * rowlen + k32 + lh * 16;
  return *(const v16h*)p;
}

#define WMMA_F16(acc, a, b)                                                    \
  acc = __builtin_amdgcn_wmma_f32_16x16x32_f16(false, (a), false, (b),         \
                                               (short)0, (acc), false, false)

static __device__ inline unsigned hashu(unsigned x) {
  x ^= x >> 16; x *= 0x7feb352dU;
  x ^= x >> 15; x *= 0x846ca68bU;
  x ^= x >> 16; return x;
}

// ---------------------------------------------------------------------------
// Kernel 1: weight prep — W1^T, W2^T, centroids to f16 (B^T layouts), |c|^2.
// ---------------------------------------------------------------------------
__global__ void mcb_prep_kernel(const float* __restrict__ W1,
                                const float* __restrict__ W2,
                                const float* __restrict__ cent,
                                _Float16* __restrict__ w1t,
                                _Float16* __restrict__ w2t,
                                _Float16* __restrict__ cf,
                                float* __restrict__ ccn) {
  int g = blockIdx.x * blockDim.x + threadIdx.x;
  int stride = gridDim.x * blockDim.x;
  for (int i = g; i < HH * FIN; i += stride) {       // w1t[n][k] = W1[k][n]
    int n = i >> 7, k = i & (FIN - 1);
    w1t[i] = (_Float16)W1[k * HH + n];
  }
  for (int i = g; i < HH * HH; i += stride) {        // w2t[n][k] = W2[k][n]
    int n = i >> 8, k = i & (HH - 1);
    w2t[i] = (_Float16)W2[k * HH + n];
  }
  for (int i = g; i < KC * HH; i += stride) cf[i] = (_Float16)cent[i];
  if (g < KC) {
    float s = 0.f;
    for (int f = 0; f < HH; ++f) { float v = cent[g * HH + f]; s += v * v; }
    ccn[g] = s;
  }
}

// ---------------------------------------------------------------------------
// Kernel 2: per-graph symmetric normalization: An = D^-1/2 (A+I) D^-1/2 (f16)
// ---------------------------------------------------------------------------
__global__ __launch_bounds__(256) void mcb_anorm_kernel(
    const float* __restrict__ adj, _Float16* __restrict__ an) {
  __shared__ float sD[NN];
  int b = blockIdx.x, t = threadIdx.x;
  const float* ag = adj + (size_t)b * NN * NN;
  if (t < NN) {
    float s = 1.f;                       // +1 self loop; clip(.,1) implied
    for (int j = 0; j < NN; ++j) s += ag[t * NN + j];
    sD[t] = rsqrtf(s);
  }
  __syncthreads();
  _Float16* og = an + (size_t)b * NN * NN;
  for (int idx = t; idx < NN * NN; idx += 256) {
    int i = idx >> 6, j = idx & 63;
    float a = ag[idx] + (i == j ? 1.f : 0.f);
    og[idx] = (_Float16)(sD[i] * a * sD[j]);
  }
}

// ---------------------------------------------------------------------------
// Kernel 3: conv1 fused.  Per block = 1 graph, 8 waves.
//  Wave w owns output strip mi = w&3, ni = (w>>2)*8 .. +7 (8 accumulators);
//  A fragment hoisted per k-step and reused across all 8 ni tiles.
//  stage1: tT = (x @ W1)^T (f16, LDS, == B^T for stage2)
//  stage2: h1 = relu(An @ t + b1) * mask  -> global f16 row-major
// ---------------------------------------------------------------------------
__global__ __launch_bounds__(256) void mcb_conv1_kernel(
    const float* __restrict__ x, const unsigned char* __restrict__ mask,
    const _Float16* __restrict__ an, const _Float16* __restrict__ w1t,
    const float* __restrict__ b1, _Float16* __restrict__ h1) {
  __shared__ __align__(32) _Float16 sA[NN * FIN];   // 16 KB
  __shared__ __align__(32) _Float16 sT[HH * NN];    // 32 KB
  __shared__ __align__(32) _Float16 sN[NN * NN];    //  8 KB
  const int b = blockIdx.x, t = threadIdx.x;
  const float* xg = x + (size_t)b * NN * FIN;
  for (int i = t; i < NN * FIN; i += 256) sA[i] = (_Float16)xg[i];
  const _Float16* ng = an + (size_t)b * NN * NN;
  for (int i = t; i < NN * NN; i += 256) sN[i] = ng[i];
  __syncthreads();

  const int w = t >> 5, l = t & 31, lr = l & 15, lh = l >> 4;
  const int mi = w & 3, nh = w >> 2;      // wave strip: mi fixed, ni = nh*8+j

  // stage 1: tT = (x @ W1)^T, K=128, 8 N-tiles per wave with shared A frag
  {
    v8f acc[8] = {};
#pragma unroll
    for (int kk = 0; kk < FIN / 32; ++kk) {
      v16h a = afrag(sA, mi * 16 + lr, FIN, kk * 32, lh);
#pragma unroll
      for (int j = 0; j < 8; ++j) {
        int ni = nh * 8 + j;
        v16h bm = bfrag(w1t, ni * 16 + lr, FIN, kk * 32, lh);
        WMMA_F16(acc[j], a, bm);
      }
    }
#pragma unroll
    for (int j = 0; j < 8; ++j) {
      int ni = nh * 8 + j;
      v8h hv;
#pragma unroll
      for (int r = 0; r < 8; ++r) hv[r] = (_Float16)acc[j][r];
      *(v8h*)(sT + (size_t)(ni * 16 + lr) * NN + mi * 16 + lh * 8) = hv;
    }
  }
  __syncthreads();

  // stage 2: h1 = relu(An @ t + b1) * mask, K=64
  {
    v8f acc[8] = {};
#pragma unroll
    for (int kk = 0; kk < NN / 32; ++kk) {
      v16h a = afrag(sN, mi * 16 + lr, NN, kk * 32, lh);
#pragma unroll
      for (int j = 0; j < 8; ++j) {
        int ni = nh * 8 + j;
        v16h bm = bfrag(sT, ni * 16 + lr, NN, kk * 32, lh);
        WMMA_F16(acc[j], a, bm);
      }
    }
    _Float16* hg = h1 + (size_t)b * NN * HH;
    const unsigned char* mg = mask + (size_t)b * NN;
#pragma unroll
    for (int j = 0; j < 8; ++j) {
      int n = (nh * 8 + j) * 16 + lr;
      float bias = b1[n];
#pragma unroll
      for (int r = 0; r < 8; ++r) {
        int m = mi * 16 + lh * 8 + r;
        float v = acc[j][r] + bias;
        v = v > 0.f ? v : 0.f;
        v = mg[m] ? v : 0.f;
        hg[(size_t)m * HH + n] = (_Float16)v;
      }
    }
  }
}

// ---------------------------------------------------------------------------
// Kernel 4: conv2 fused. A operand (h1, f16 row-major) streamed from global,
//  hoisted per k-step across the wave's 8 N-tiles (8x fewer global A loads).
//  h2 written both f32 (pooling / |h|^2) and f16 (centroid WMMA A-frags).
// ---------------------------------------------------------------------------
__global__ __launch_bounds__(256) void mcb_conv2_kernel(
    const _Float16* __restrict__ h1, const unsigned char* __restrict__ mask,
    const _Float16* __restrict__ an, const _Float16* __restrict__ w2t,
    const float* __restrict__ b2, float* __restrict__ h2f,
    _Float16* __restrict__ h2h) {
  __shared__ __align__(32) _Float16 sT[HH * NN];    // 32 KB
  __shared__ __align__(32) _Float16 sN[NN * NN];    //  8 KB
  const int b = blockIdx.x, t = threadIdx.x;
  const _Float16* ng = an + (size_t)b * NN * NN;
  for (int i = t; i < NN * NN; i += 256) sN[i] = ng[i];
  __syncthreads();

  const int w = t >> 5, l = t & 31, lr = l & 15, lh = l >> 4;
  const int mi = w & 3, nh = w >> 2;
  const _Float16* ag = h1 + (size_t)b * NN * HH;

  // stage 1: tT = (h1 @ W2)^T, K=256
  {
    v8f acc[8] = {};
#pragma unroll
    for (int kk = 0; kk < HH / 32; ++kk) {
      v16h a = afrag(ag, mi * 16 + lr, HH, kk * 32, lh);
#pragma unroll
      for (int j = 0; j < 8; ++j) {
        int ni = nh * 8 + j;
        v16h bm = bfrag(w2t, ni * 16 + lr, HH, kk * 32, lh);
        WMMA_F16(acc[j], a, bm);
      }
    }
#pragma unroll
    for (int j = 0; j < 8; ++j) {
      int ni = nh * 8 + j;
      v8h hv;
#pragma unroll
      for (int r = 0; r < 8; ++r) hv[r] = (_Float16)acc[j][r];
      *(v8h*)(sT + (size_t)(ni * 16 + lr) * NN + mi * 16 + lh * 8) = hv;
    }
  }
  __syncthreads();

  // stage 2: h2 = relu(An @ t2 + b2) * mask
  {
    v8f acc[8] = {};
#pragma unroll
    for (int kk = 0; kk < NN / 32; ++kk) {
      v16h a = afrag(sN, mi * 16 + lr, NN, kk * 32, lh);
#pragma unroll
      for (int j = 0; j < 8; ++j) {
        int ni = nh * 8 + j;
        v16h bm = bfrag(sT, ni * 16 + lr, NN, kk * 32, lh);
        WMMA_F16(acc[j], a, bm);
      }
    }
    float*    og = h2f + (size_t)b * NN * HH;
    _Float16* oh = h2h + (size_t)b * NN * HH;
    const unsigned char* mg = mask + (size_t)b * NN;
#pragma unroll
    for (int j = 0; j < 8; ++j) {
      int n = (nh * 8 + j) * 16 + lr;
      float bias = b2[n];
#pragma unroll
      for (int r = 0; r < 8; ++r) {
        int m = mi * 16 + lh * 8 + r;
        float v = acc[j][r] + bias;
        v = v > 0.f ? v : 0.f;
        v = mg[m] ? v : 0.f;
        og[(size_t)m * HH + n] = v;
        oh[(size_t)m * HH + n] = (_Float16)v;
      }
    }
  }
}

// ---------------------------------------------------------------------------
// Kernel 5: distances (WMMA h.c^T) + softmin + Gumbel-max MC sampling +
//           masked product -> graph_prob.  One block per graph.
// ---------------------------------------------------------------------------
__global__ __launch_bounds__(256) void mcb_dist_kernel(
    const float* __restrict__ h2f, const _Float16* __restrict__ h2h,
    const _Float16* __restrict__ cf, const float* __restrict__ ccn,
    const unsigned char* __restrict__ mask, int* __restrict__ concepts,
    float* __restrict__ gprob) {
  __shared__ float sDot[NN * KC];     // 8 KB
  __shared__ float sHHp[NN][4];
  __shared__ float sH2[NN];
  __shared__ float sNP[SMC][NN];
  const int b = blockIdx.x, t = threadIdx.x;
  const float* hg = h2f + (size_t)b * NN * HH;

  // |h|^2 per node (partials over 4 quarters)
  {
    int m = t >> 2, q = t & 3;
    const float* hr = hg + (size_t)m * HH + q * 64;
    float s = 0.f;
    for (int f = 0; f < 64; ++f) { float v = hr[f]; s += v * v; }
    sHHp[m][q] = s;
  }
  __syncthreads();
  if (t < NN) sH2[t] = sHHp[t][0] + sHHp[t][1] + sHHp[t][2] + sHHp[t][3];

  // dot = h2 @ c^T : [64 x 32], 4x2 tiles, one tile per wave, K=256
  const int w = t >> 5, l = t & 31, lr = l & 15, lh = l >> 4;
  {
    int mi = w & 3, ni = w >> 2;
    v8f acc = {};
    const _Float16* ag = h2h + (size_t)b * NN * HH;
#pragma unroll
    for (int kk = 0; kk < HH / 32; ++kk) {
      v16h a  = afrag(ag, mi * 16 + lr, HH, kk * 32, lh);
      v16h bm = bfrag(cf, ni * 16 + lr, HH, kk * 32, lh);
      WMMA_F16(acc, a, bm);
    }
#pragma unroll
    for (int r = 0; r < 8; ++r)
      sDot[(mi * 16 + lh * 8 + r) * KC + ni * 16 + lr] = acc[r];
  }
  __syncthreads();

  // per-node softmin over 32 clusters (lane == cluster), Gumbel-max sampling
  for (int i = 0; i < 8; ++i) {
    int m = w * 8 + i;
    float d2 = sH2[m] + ccn[l] - 2.f * sDot[m * KC + l];
    float d = sqrtf(fmaxf(d2, 1e-12f));
    float logit = -10.f * d;                       // -dist / TEMP
    float mx = logit;
    for (int o = 16; o > 0; o >>= 1) mx = fmaxf(mx, __shfl_xor(mx, o, 32));
    float e = expf(logit - mx);
    float se = e;
    for (int o = 16; o > 0; o >>= 1) se += __shfl_xor(se, o, 32);
    float prob = e / se;
    float logp = (logit - mx) - logf(se);
    unsigned gid = (unsigned)(b * NN + m);
    for (int s = 0; s < SMC; ++s) {
      unsigned h = hashu(((unsigned)(s + 1) * 0x9E3779B9u) ^
                         (gid * 0x85EBCA6Bu) ^ ((unsigned)l * 0xC2B2AE35u));
      float u = (float)(h & 0xFFFFFFu) * (1.0f / 16777216.0f);
      u = fminf(fmaxf(u, 1e-7f), 1.f - 1e-7f);
      float g = -logf(-logf(u));
      float bv = logp + g;
      int bi = l;
      for (int o = 16; o > 0; o >>= 1) {
        float ov = __shfl_xor(bv, o, 32);
        int   oi = __shfl_xor(bi, o, 32);
        if (ov > bv || (ov == bv && oi < bi)) { bv = ov; bi = oi; }
      }
      float np = __shfl(prob, bi, 32);
      if (l == 0) {
        concepts[((size_t)s * NB + b) * NN + m] = bi;
        sNP[s][m] = np;
      }
    }
  }
  __syncthreads();
  if (t < SMC) {
    float p = 1.f;
    const unsigned char* mg = mask + (size_t)b * NN;
    for (int m = 0; m < NN; ++m)
      if (mg[m]) p *= sNP[t][m];
    gprob[(size_t)t * NB + b] = p;
  }
}

// ---------------------------------------------------------------------------
// Kernel 6: connected components (min-label propagation, N iters like the
//  reference scan) on concept-restricted adjacency bitmasks, then scatter-mean
//  pooling of h2 and S^T A S binarized.  One block per (s,b) sample graph.
// ---------------------------------------------------------------------------
__global__ __launch_bounds__(64) void mcb_cc_kernel(
    const float* __restrict__ adj, const unsigned char* __restrict__ mask,
    const int* __restrict__ concepts, const float* __restrict__ h2f,
    float* __restrict__ newx, float* __restrict__ newadj,
    float* __restrict__ labout) {
  __shared__ int sCon[NN];
  __shared__ unsigned char sMk[NN];
  __shared__ unsigned sOb[NN][2];     // original (masked) adjacency bits
  __shared__ int sLab[NN];
  __shared__ int sLabF[NN];
  __shared__ unsigned sMem[NN][2];    // cluster membership bits
  __shared__ int sCnt[NN];

  const int sb = blockIdx.x;
  const int b = sb & (NB - 1);
  const int i = threadIdx.x;
  const float* ag = adj + (size_t)b * NN * NN;

  sCon[i] = concepts[(size_t)sb * NN + i];
  sMk[i]  = mask[(size_t)b * NN + i];
  unsigned lo = 0, hi = 0;
  for (int j = 0; j < NN; ++j) {
    if (ag[i * NN + j] > 0.f) {
      if (j < 32) lo |= 1u << j; else hi |= 1u << (j - 32);
    }
  }
  sOb[i][0] = lo; sOb[i][1] = hi;
  __syncthreads();

  // restrict to same-concept edges
  unsigned clo = 0, chi = 0;
  {
    int ci = sCon[i];
    unsigned m0 = lo;
    while (m0) { int j = __builtin_ctz(m0); m0 &= m0 - 1;
                 if (sCon[j] == ci) clo |= 1u << j; }
    unsigned m1 = hi;
    while (m1) { int j = __builtin_ctz(m1); m1 &= m1 - 1;
                 if (sCon[j + 32] == ci) chi |= 1u << j; }
  }
  sLab[i] = i + 1;
  __syncthreads();

  for (int it = 0; it < NN; ++it) {
    int mn = sLab[i];
    unsigned m0 = clo;
    while (m0) { int j = __builtin_ctz(m0); m0 &= m0 - 1;
                 int v = sLab[j]; mn = v < mn ? v : mn; }
    unsigned m1 = chi;
    while (m1) { int j = __builtin_ctz(m1); m1 &= m1 - 1;
                 int v = sLab[j + 32]; mn = v < mn ? v : mn; }
    __syncthreads();
    sLab[i] = mn;
    __syncthreads();
  }

  int labv = sMk[i] ? sLab[i] : 0;
  labout[(size_t)sb * NN + i] = (float)labv;
  sLabF[i] = labv;
  __syncthreads();

  // membership bits for cluster c=i (label i+1); cluster 0 (masked) dropped
  unsigned mlo = 0, mhi = 0; int cnt = 0;
  for (int j = 0; j < NN; ++j) {
    if (sLabF[j] == i + 1) {
      if (j < 32) mlo |= 1u << j; else mhi |= 1u << (j - 32);
      ++cnt;
    }
  }
  sMem[i][0] = mlo; sMem[i][1] = mhi; sCnt[i] = cnt;
  __syncthreads();

  // new_adj[c][d] = any edge between members_c and members_d
  unsigned rlo = 0, rhi = 0;
  {
    unsigned m0 = mlo;
    while (m0) { int j = __builtin_ctz(m0); m0 &= m0 - 1;
                 rlo |= sOb[j][0]; rhi |= sOb[j][1]; }
    unsigned m1 = mhi;
    while (m1) { int j = __builtin_ctz(m1); m1 &= m1 - 1;
                 rlo |= sOb[j + 32][0]; rhi |= sOb[j + 32][1]; }
  }
  float* nar = newadj + (size_t)sb * NN * NN + (size_t)i * NN;
  for (int d = 0; d < NN; ++d) {
    unsigned inter = (rlo & sMem[d][0]) | (rhi & sMem[d][1]);
    nar[d] = inter ? 1.0f : 0.0f;
  }

  // new_x[c][f] = mean over member nodes of h2 (0 if empty cluster)
  const float* hg = h2f + (size_t)b * NN * HH;
  float* nx = newx + (size_t)sb * NN * HH;
  for (int idx = i; idx < NN * HH; idx += NN) {
    int c = idx >> 8, f = idx & (HH - 1);
    float s = 0.f;
    int cn = sCnt[c];
    if (cn > 0) {
      unsigned m0 = sMem[c][0];
      while (m0) { int j = __builtin_ctz(m0); m0 &= m0 - 1;
                   s += hg[(size_t)j * HH + f]; }
      unsigned m1 = sMem[c][1];
      while (m1) { int j = __builtin_ctz(m1); m1 &= m1 - 1;
                   s += hg[(size_t)(j + 32) * HH + f]; }
      s /= (float)cn;
    }
    nx[idx] = s;
  }
}

// ---------------------------------------------------------------------------
// Launch
// ---------------------------------------------------------------------------
extern "C" void kernel_launch(void* const* d_in, const int* in_sizes, int n_in,
                              void* d_out, int out_size, void* d_ws,
                              size_t ws_size, hipStream_t stream) {
  const float*         x    = (const float*)d_in[0];
  const float*         adj  = (const float*)d_in[1];
  const unsigned char* mask = (const unsigned char*)d_in[2];
  const float*         W1   = (const float*)d_in[3];
  const float*         b1   = (const float*)d_in[4];
  const float*         W2   = (const float*)d_in[5];
  const float*         b2   = (const float*)d_in[6];
  const float*         cent = (const float*)d_in[7];

  char* ws = (char*)d_ws;
  size_t off = 0;
  auto carve = [&](size_t bytes) -> char* {
    char* p = ws + off;
    off += (bytes + 255) & ~(size_t)255;
    return p;
  };
  _Float16* w1t   = (_Float16*)carve((size_t)HH * FIN * 2);       // 64 KB
  _Float16* w2t   = (_Float16*)carve((size_t)HH * HH * 2);        // 128 KB
  _Float16* cf16  = (_Float16*)carve((size_t)KC * HH * 2);        // 16 KB
  float*    ccn   = (float*)carve((size_t)KC * 4);
  _Float16* anorm = (_Float16*)carve((size_t)NB * NN * NN * 2);   // 4 MB
  _Float16* h1    = (_Float16*)carve((size_t)NB * NN * HH * 2);   // 16 MB
  float*    h2f   = (float*)carve((size_t)NB * NN * HH * 4);      // 33.5 MB
  _Float16* h2h   = (_Float16*)carve((size_t)NB * NN * HH * 2);   // 16 MB
  int*      conc  = (int*)carve((size_t)SMC * NB * NN * 4);       // 0.5 MB
  (void)ws_size; (void)in_sizes; (void)n_in; (void)out_size;

  float* out = (float*)d_out;
  const size_t OUT_NEWX   = 0;
  const size_t OUT_NEWADJ = (size_t)SMC * NB * NN * HH;           // 33554432
  const size_t OUT_GP     = OUT_NEWADJ + (size_t)SMC * NB * NN * NN;
  const size_t OUT_LAB    = OUT_GP + (size_t)SMC * NB;

  mcb_prep_kernel<<<128, 256, 0, stream>>>(W1, W2, cent, w1t, w2t, cf16, ccn);
  mcb_anorm_kernel<<<NB, 256, 0, stream>>>(adj, anorm);
  mcb_conv1_kernel<<<NB, 256, 0, stream>>>(x, mask, anorm, w1t, b1, h1);
  mcb_conv2_kernel<<<NB, 256, 0, stream>>>(h1, mask, anorm, w2t, b2, h2f, h2h);
  mcb_dist_kernel<<<NB, 256, 0, stream>>>(h2f, h2h, cf16, ccn, mask, conc,
                                          out + OUT_GP);
  mcb_cc_kernel<<<SMC * NB, 64, 0, stream>>>(adj, mask, conc, h2f,
                                             out + OUT_NEWX, out + OUT_NEWADJ,
                                             out + OUT_LAB);
}
